// GCNSimple_70463233458671
// MI455X (gfx1250) — matrix operand order repaired and checked by reading, not compiled, not run on hardware
//
#include <hip/hip_runtime.h>
#include <hip/hip_bf16.h>

typedef __attribute__((ext_vector_type(16))) __bf16 v16bf;
typedef __attribute__((ext_vector_type(8)))  float  v8f;

#define NNODES 100000
#define NEDGES 1600000
#define F_IN   128
#define F_HID  64
#define F_OUT  40
#define BN_EPS 1e-5f

__device__ __forceinline__ __bf16 f2bf(float f) { return (__bf16)f; }

// ---------------- degree / norm ----------------
__global__ void deg_init(float* __restrict__ deg) {
    int i = blockIdx.x * blockDim.x + threadIdx.x;
    if (i < NNODES) deg[i] = 1.0f;                 // self-loop contributes 1
}

__global__ void deg_count(const int* __restrict__ cols, float* __restrict__ deg) {
    int e = blockIdx.x * blockDim.x + threadIdx.x;
    if (e < NEDGES) atomicAdd(&deg[cols[e]], 1.0f);
}

__global__ void deg_finish(float* __restrict__ deg) {
    int i = blockIdx.x * blockDim.x + threadIdx.x;
    if (i < NNODES) { float d = deg[i]; deg[i] = (d > 0.f) ? rsqrtf(d) : 0.f; }
}

// ---------------- BN fold ----------------
__global__ void bn_prep(const float* __restrict__ gamma, const float* __restrict__ beta,
                        const float* __restrict__ mean,  const float* __restrict__ var,
                        float* __restrict__ sc, float* __restrict__ sh) {
    int i = threadIdx.x;
    if (i < F_IN) {
        float s = gamma[i] * rsqrtf(var[i] + BN_EPS);
        sc[i] = s;
        sh[i] = beta[i] - mean[i] * s;
    }
}

// ---------------- GEMM1: h = BN(x) @ W1   [100000x128]x[128x64] ----------------
__global__ void __launch_bounds__(256) bn_gemm1(const float* __restrict__ x,
                                                const float* __restrict__ sc,
                                                const float* __restrict__ sh,
                                                const float* __restrict__ W1,
                                                float* __restrict__ h) {
    const int lane = threadIdx.x & 31;
    const int gw   = (blockIdx.x * blockDim.x + threadIdx.x) >> 5;
    const int nw   = (gridDim.x * blockDim.x) >> 5;
    const int hi   = lane >> 4;        // which half-wave
    const int nB   = lane & 15;        // B/C/D column owned by this lane
    const int m    = lane & 15;        // A row owned by this lane

    // B fragments for W1 (row-major [k][n]); same k order as A: lanes 0-15 K=0..7,16..23, lanes 16-31 K=8..15,24..31
    v16bf bfrag[4][4];
    for (int s = 0; s < 4; ++s)
        for (int nt = 0; nt < 4; ++nt) {
            int n  = nt * 16 + nB;
            int kb = s * 32 + hi * 8;
            #pragma unroll
            for (int j = 0; j < 16; ++j)
                bfrag[s][nt][j] = f2bf(W1[(kb + ((j < 8) ? j : (8 + j))) * F_HID + n]);
        }

    const int kAb   = hi * 8;          // A layout: lanes 0-15 kbase=0, lanes 16-31 kbase=8
    const int ntile = NNODES / 16;     // 6250 exactly
    for (int t = gw; t < ntile; t += nw) {
        const int row = t * 16 + m;
        const float* xr = x + (size_t)row * F_IN;
        v8f acc[4] = {};
        for (int s = 0; s < 4; ++s) {
            v16bf a;
            #pragma unroll
            for (int j = 0; j < 16; ++j) {
                int k = kAb + ((j < 8) ? j : (8 + j));   // ISA 16-bit A 16x32 layout
                int f = s * 32 + k;
                a[j] = f2bf(xr[f] * sc[f] + sh[f]);      // fused BatchNorm
            }
            for (int nt = 0; nt < 4; ++nt)
                acc[nt] = __builtin_amdgcn_wmma_f32_16x16x32_bf16(
                    false, a, false, bfrag[s][nt], (short)0, acc[nt], false, false);
        }
        const int mb = t * 16 + hi * 8;                  // C/D: VGPR r -> row mb+r
        for (int nt = 0; nt < 4; ++nt) {
            int n = nt * 16 + nB;
            #pragma unroll
            for (int r = 0; r < 8; ++r)
                h[(size_t)(mb + r) * F_HID + n] = acc[nt][r];
        }
    }
}

// ---------------- GEMM2: g = relu(agg1 + b1) @ W2   [100000x64]x[64x40] ----------------
__global__ void __launch_bounds__(256) relu_gemm2(const float* __restrict__ agg1,
                                                  const float* __restrict__ b1,
                                                  const float* __restrict__ W2,
                                                  float* __restrict__ g) {
    const int lane = threadIdx.x & 31;
    const int gw   = (blockIdx.x * blockDim.x + threadIdx.x) >> 5;
    const int nw   = (gridDim.x * blockDim.x) >> 5;
    const int hi   = lane >> 4;
    const int nB   = lane & 15;
    const int m    = lane & 15;

    v16bf bfrag[2][3];                 // K=64 -> 2 steps; N=40 padded to 48 -> 3 tiles
    for (int s = 0; s < 2; ++s)
        for (int nt = 0; nt < 3; ++nt) {
            int n  = nt * 16 + nB;
            int kb = s * 32 + hi * 8;
            #pragma unroll
            for (int j = 0; j < 16; ++j)
                bfrag[s][nt][j] = f2bf((n < F_OUT) ? W2[(kb + ((j < 8) ? j : (8 + j))) * F_OUT + n] : 0.f);
        }

    const int kAb   = hi * 8;
    const int ntile = NNODES / 16;
    for (int t = gw; t < ntile; t += nw) {
        const int row = t * 16 + m;
        const float* ar = agg1 + (size_t)row * F_HID;
        v8f acc[3] = {};
        for (int s = 0; s < 2; ++s) {
            v16bf a;
            #pragma unroll
            for (int j = 0; j < 16; ++j) {
                int k = kAb + ((j < 8) ? j : (8 + j));
                int f = s * 32 + k;
                a[j] = f2bf(fmaxf(ar[f] + b1[f], 0.f));  // fused bias + ReLU
            }
            for (int nt = 0; nt < 3; ++nt)
                acc[nt] = __builtin_amdgcn_wmma_f32_16x16x32_bf16(
                    false, a, false, bfrag[s][nt], (short)0, acc[nt], false, false);
        }
        const int mb = t * 16 + hi * 8;
        for (int nt = 0; nt < 3; ++nt) {
            int n = nt * 16 + nB;
            if (n < F_OUT) {
                #pragma unroll
                for (int r = 0; r < 8; ++r)
                    g[(size_t)(mb + r) * F_OUT + n] = acc[nt][r];
            }
        }
    }
}

// ---------------- normalized scatter-add: dst[col] += src[row] * dinv[row]*dinv[col] ----------------
__global__ void agg_scatter(const float* __restrict__ src, const int* __restrict__ rows,
                            const int* __restrict__ cols, const float* __restrict__ dinv,
                            float* __restrict__ dst, int F) {
    const int lane = threadIdx.x & 31;
    const long long gw = ((long long)blockIdx.x * blockDim.x + threadIdx.x) >> 5;
    const long long total = (long long)NEDGES + NNODES;
    if (gw >= total) return;
    int r, c;
    if (gw < NEDGES) { r = rows[gw]; c = cols[gw]; }
    else             { r = c = (int)(gw - NEDGES); }       // self-loop
    float nrm = dinv[r] * dinv[c];
    const float* sp = src + (size_t)r * F;
    float*       dp = dst + (size_t)c * F;
    for (int f = lane; f < F; f += 32)
        atomicAdd(dp + f, sp[f] * nrm);
}

// ---------------- log_softmax over 40 features (+b2), in place, one wave per node ----------------
__global__ void logsoftmax40(float* __restrict__ out, const float* __restrict__ b2) {
    const int lane = threadIdx.x & 31;
    const int node = (blockIdx.x * blockDim.x + threadIdx.x) >> 5;
    if (node >= NNODES) return;
    float* p = out + (size_t)node * F_OUT;
    float v0 = p[lane] + b2[lane];
    float v1 = (lane < F_OUT - 32) ? (p[32 + lane] + b2[32 + lane]) : -__builtin_inff();
    float mx = fmaxf(v0, v1);
    #pragma unroll
    for (int off = 16; off > 0; off >>= 1) mx = fmaxf(mx, __shfl_xor(mx, off, 32));
    float s = __expf(v0 - mx) + ((lane < F_OUT - 32) ? __expf(v1 - mx) : 0.f);
    #pragma unroll
    for (int off = 16; off > 0; off >>= 1) s += __shfl_xor(s, off, 32);
    float lse = mx + __logf(s);
    p[lane] = v0 - lse;
    if (lane < F_OUT - 32) p[32 + lane] = v1 - lse;
}

extern "C" void kernel_launch(void* const* d_in, const int* in_sizes, int n_in,
                              void* d_out, int out_size, void* d_ws, size_t ws_size,
                              hipStream_t stream) {
    (void)in_sizes; (void)n_in; (void)out_size; (void)ws_size;
    const float* x     = (const float*)d_in[0];
    const int*   ei    = (const int*)d_in[1];   // [2,E] int32 (JAX demotes int64 w/o x64)
    const float* gamma = (const float*)d_in[2];
    const float* beta  = (const float*)d_in[3];
    const float* rmean = (const float*)d_in[4];
    const float* rvar  = (const float*)d_in[5];
    const float* W1    = (const float*)d_in[6];
    const float* b1    = (const float*)d_in[7];
    const float* W2    = (const float*)d_in[8];
    const float* b2    = (const float*)d_in[9];
    float* out = (float*)d_out;
    float* ws  = (float*)d_ws;

    const int* rows = ei;            // edge_index[0]
    const int* cols = ei + NEDGES;   // edge_index[1]

    // workspace layout (floats): total ~16.9M floats (~68 MB)
    float* dinv = ws;                                  // N (deg -> dinv in place)
    float* bnsc = ws + 100352;                         // 128
    float* bnsh = bnsc + 128;                          // 128
    float* h    = ws + 101376;                         // N*64
    float* agg1 = h + (size_t)NNODES * F_HID;          // N*64
    float* g    = agg1 + (size_t)NNODES * F_HID;       // N*40

    // degree / normalization
    deg_init  <<<(NNODES + 255) / 256, 256, 0, stream>>>(dinv);
    deg_count <<<(NEDGES + 255) / 256, 256, 0, stream>>>(cols, dinv);
    deg_finish<<<(NNODES + 255) / 256, 256, 0, stream>>>(dinv);

    // BN fold
    bn_prep<<<1, 128, 0, stream>>>(gamma, beta, rmean, rvar, bnsc, bnsh);

    // zero accumulators (fresh every call; graph-capturable)
    hipMemsetAsync(agg1, 0, (size_t)NNODES * F_HID * sizeof(float), stream);
    hipMemsetAsync(out,  0, (size_t)NNODES * F_OUT * sizeof(float), stream);

    // layer 1: WMMA GEMM then scatter
    bn_gemm1<<<782, 256, 0, stream>>>(x, bnsc, bnsh, W1, h);  // 782*8 waves >= 6250 tiles
    const long long tot = (long long)NEDGES + NNODES;
    const int sblocks = (int)((tot * 32 + 255) / 256);
    agg_scatter<<<sblocks, 256, 0, stream>>>(h, rows, cols, dinv, agg1, F_HID);

    // layer 2: WMMA GEMM then scatter directly into d_out
    relu_gemm2<<<782, 256, 0, stream>>>(agg1, b1, W2, g);
    agg_scatter<<<sblocks, 256, 0, stream>>>(g, rows, cols, dinv, out, F_OUT);

    // log-softmax in place
    logsoftmax40<<<(NNODES * 32 + 255) / 256, 256, 0, stream>>>(out, b2);
}
